// SceneNarrativeEngine_72670846648628
// MI455X (gfx1250) — matrix operand order repaired and checked by reading, not compiled, not run on hardware
//
#include <hip/hip_runtime.h>
#include <hip/hip_bf16.h>
#include <math.h>

// ---------------- CDNA5 WMMA types ----------------
typedef __attribute__((ext_vector_type(16))) __bf16 v16bf;
typedef __attribute__((ext_vector_type(8)))  float  v8f;

union BFrag {
    v16bf v;
    unsigned short u[16];
    uint4 q[2];
};

__device__ __forceinline__ unsigned short f2bf(float f) {
    union { float f; unsigned int u; } cv;
    cv.f = f;
    unsigned int r = cv.u + 0x7FFFu + ((cv.u >> 16) & 1u); // round-to-nearest-even
    return (unsigned short)(r >> 16);
}

__device__ __forceinline__ float sigmoidf_(float x) { return 1.0f / (1.0f + __expf(-x)); }
__device__ __forceinline__ float softplusf_(float x) { return (x > 20.0f) ? x : log1pf(__expf(x)); }
__device__ __forceinline__ float geluf_(float x) { return 0.5f * x * (1.0f + erff(x * 0.70710678118654752f)); }

// ---------------- constants ----------------
#define NBS 4
#define NNS 512
#define NNU 32
#define DM1 1024
#define DD 256
#define DINNER 512
#define DSTATE 16
#define DTRANK 16
#define NSLOTS 50
#define NTOK (NBS * NNS * NNU)   // 65536
#define NROW (NBS * NNS)         // 2048

// =====================================================================
// Kernel: pre-swizzle sg_w1 (1024x512 f32) into bf16 WMMA B-fragment layout.
// Layout: Bswz[nt][kc][lane][e], nt=0..31 (16-wide N tiles), kc=0..31 (32-wide K chunks)
// B fragment (16-bit, 32x16): lane l -> col n = nt*16 + l%16, element e -> k = kc*32 + 16*(l/16) + e
// =====================================================================
__global__ __launch_bounds__(256) void swz_w1_kernel(const float* __restrict__ W1,
                                                     unsigned short* __restrict__ Bswz) {
    int g = blockIdx.x * 256 + threadIdx.x;       // 0 .. 524287
    int e  = g & 15;
    int l  = (g >> 4) & 31;
    int kc = (g >> 9) & 31;
    int nt = g >> 14;                             // 0..31
    int n  = nt * 16 + (l & 15);
    int h  = l >> 4;
    int k  = kc * 32 + 16 * h + e;
    Bswz[g] = f2bf(W1[k * 512 + n]);
}

// =====================================================================
// Kernel: kq[d] = key_w[d,:] . query ; kq[1024] = key_b . query
// =====================================================================
__global__ __launch_bounds__(256) void kq_kernel(const float* __restrict__ key_w,
                                                 const float* __restrict__ key_b,
                                                 const float* __restrict__ query,
                                                 float* __restrict__ kq) {
    int d = blockIdx.x * 256 + threadIdx.x;
    if (d < 1024) {
        float acc = 0.0f;
        for (int j = 0; j < 1024; ++j) acc += key_w[d * 1024 + j] * query[j];
        kq[d] = acc;
    } else if (d == 1024) {
        float acc = 0.0f;
        for (int j = 0; j < 1024; ++j) acc += key_b[j] * query[j];
        kq[1024] = acc;
    }
}

// =====================================================================
// Big WMMA kernel: per block computes 32 tokens x (K=1024 -> N=512) GEMM in
// bf16 WMMA, fuses +b1, GELU, .w2, +b2, sigmoid -> s[token];
// also fuses rawdot[token] = x_token . kq from the staged registers.
//
// A tile is converted to bf16 ONCE by the 256 staging threads and scattered
// directly into WMMA A-fragment layout in LDS (double-buffered, one barrier
// per K-chunk). Each wave's A fragment = two contiguous ds_load_b128.
// 8 waves: wave w -> row subtile r=w&1 (16 rows), col group cg=w>>1 (128 cols)
// =====================================================================
__global__ __launch_bounds__(256) void gemm_gate_kernel(const float* __restrict__ X,
                                                        const unsigned short* __restrict__ Bswz,
                                                        const float* __restrict__ b1,
                                                        const float* __restrict__ w2,
                                                        const float* __restrict__ b2,
                                                        const float* __restrict__ kq,
                                                        float* __restrict__ s_out,
                                                        float* __restrict__ dot_out) {
    // [buf][row-subtile][lane][element] : bf16 A fragments, 4 KB total
    __shared__ __align__(16) unsigned short abf[2][2][32][16];
    __shared__ float xpart[32][8];
    __shared__ float rowsum[32][4];

    const int tid  = threadIdx.x;
    const int wave = tid >> 5;
    const int lane = tid & 31;
    const int r    = wave & 1;       // row subtile (16 rows)
    const int cg   = wave >> 1;      // col group (128 cols)
    const int l16  = lane & 15;
    const int h    = lane >> 4;
    const int m0   = blockIdx.x * 32;

    const int sRow = tid >> 3;       // staging: row 0..31
    const int sK4  = (tid & 7) * 4;  // staging: k base 0,4,...,28
    const int rsub = sRow >> 4;
    const int lrow = sRow & 15;

    v8f zero8 = {0.f, 0.f, 0.f, 0.f, 0.f, 0.f, 0.f, 0.f};
    v8f acc[8];
#pragma unroll
    for (int t = 0; t < 8; ++t) acc[t] = zero8;

    float xdot = 0.0f;

    for (int kc = 0; kc < 32; ++kc) {
        // ---- stage: coalesced float4 load, fused x.kq partial, bf16 scatter ----
        const float4 av = *reinterpret_cast<const float4*>(
            X + (size_t)(m0 + sRow) * 1024 + kc * 32 + sK4);
        if (kc < 31) {
            __builtin_prefetch(X + (size_t)(m0 + sRow) * 1024 + (kc + 1) * 32 + sK4, 0, 3);
        }
        float va[4] = {av.x, av.y, av.z, av.w};
#pragma unroll
        for (int j = 0; j < 4; ++j) {
            const int k  = sK4 + j;
            const int hh = k >> 4;                       // lane half
            const int e  = k & 15;                       // fragment element
            abf[kc & 1][rsub][16 * hh + lrow][e] = f2bf(va[j]);
            xdot += va[j] * kq[kc * 32 + k];
        }
        __syncthreads();   // single barrier per chunk (double-buffered tile)

        // ---- A fragment: two contiguous 16B LDS loads, no conversion ----
        BFrag af;
        const uint4* ap = reinterpret_cast<const uint4*>(&abf[kc & 1][r][lane][0]);
        af.q[0] = ap[0];
        af.q[1] = ap[1];

        // ---- 8 B fragments (coalesced global b128 pairs) + WMMAs ----
#pragma unroll
        for (int t = 0; t < 8; ++t) {
            const int nt = cg * 8 + t;
            BFrag bf;
            const uint4* bp = reinterpret_cast<const uint4*>(
                Bswz + ((size_t)(nt * 32 + kc) * 32 + lane) * 16);
            bf.q[0] = bp[0];
            bf.q[1] = bp[1];
            acc[t] = __builtin_amdgcn_wmma_f32_16x16x32_bf16(
                false, af.v, false, bf.v, (short)0, acc[t], false, false);
        }
    }

    // reduce fused x.kq dot
    xpart[sRow][tid & 7] = xdot;
    __syncthreads();
    if (tid < 32) {
        float d = 0.0f;
#pragma unroll
        for (int q = 0; q < 8; ++q) d += xpart[tid][q];
        dot_out[m0 + tid] = d;
    }

    // epilogue: +b1, GELU, * w2, row-reduce
    float rowpart[8] = {0.f, 0.f, 0.f, 0.f, 0.f, 0.f, 0.f, 0.f};
#pragma unroll
    for (int t = 0; t < 8; ++t) {
        const int col = cg * 128 + t * 16 + l16;
        const float bb = b1[col];
        const float ww = w2[col];
#pragma unroll
        for (int v = 0; v < 8; ++v) {
            float g = acc[t][v] + bb;
            rowpart[v] += geluf_(g) * ww;
        }
    }
#pragma unroll
    for (int v = 0; v < 8; ++v) {
        float rv = rowpart[v];
        rv += __shfl_xor(rv, 1, 32);
        rv += __shfl_xor(rv, 2, 32);
        rv += __shfl_xor(rv, 4, 32);
        rv += __shfl_xor(rv, 8, 32);
        rowpart[v] = rv;
    }
    if (l16 == 0) {
#pragma unroll
        for (int v = 0; v < 8; ++v) rowsum[16 * r + v + 8 * h][cg] = rowpart[v];
    }
    __syncthreads();
    if (tid < 32) {
        float tot = rowsum[tid][0] + rowsum[tid][1] + rowsum[tid][2] + rowsum[tid][3] + b2[0];
        s_out[m0 + tid] = sigmoidf_(tot);
    }
}

// =====================================================================
// Per-scene softmax with decay penalty + mask -> weight w = alpha * s
// one wave (32 lanes = 32 tokens) per scene
// =====================================================================
__global__ __launch_bounds__(32) void pool_softmax_kernel(const float* __restrict__ s,
                                                          const float* __restrict__ dot,
                                                          const unsigned char* __restrict__ mask,
                                                          const float* __restrict__ kq,
                                                          const float* __restrict__ decay,
                                                          float* __restrict__ wout) {
    int sc = blockIdx.x;
    int t = threadIdx.x;
    int tok = sc * 32 + t;
    bool m = mask[tok] != 0;

    float valid = m ? 0.0f : 1.0f;
    for (int o = 16; o > 0; o >>= 1) valid += __shfl_xor(valid, o, 32);

    float sv = s[tok];
    float logit = (sv * dot[tok] + kq[1024]) * (1.0f / 32.0f);
    float lam = softplusf_(decay[0]);
    float pen = fmaxf(valid - 1.0f - (float)t, 0.0f);
    logit -= lam * pen;
    if (m) logit = -10000.0f;

    float mx = logit;
    for (int o = 16; o > 0; o >>= 1) mx = fmaxf(mx, __shfl_xor(mx, o, 32));
    float e = __expf(logit - mx);
    float se = e;
    for (int o = 16; o > 0; o >>= 1) se += __shfl_xor(se, o, 32);
    wout[tok] = (e / se) * sv;
}

// =====================================================================
// scene[sc,:] = sum_t w[sc,t] * x[sc,t,:]
// =====================================================================
__global__ __launch_bounds__(256) void scene_sum_kernel(const float* __restrict__ X,
                                                        const float* __restrict__ w,
                                                        float* __restrict__ scene) {
    __shared__ float lw[32];
    int sc = blockIdx.x;
    int tid = threadIdx.x;
    if (tid < 32) lw[tid] = w[sc * 32 + tid];
    __syncthreads();
    float acc[4] = {0.f, 0.f, 0.f, 0.f};
    const float* xb = X + (size_t)sc * 32 * 1024;
    for (int t = 0; t < 32; ++t) {
        float wt = lw[t];
#pragma unroll
        for (int q = 0; q < 4; ++q) acc[q] += wt * xb[t * 1024 + q * 256 + tid];
    }
#pragma unroll
    for (int q = 0; q < 4; ++q) scene[(size_t)sc * 1024 + q * 256 + tid] = acc[q];
}

// =====================================================================
// h = scene @ proj_w + proj_b  (2048x1024 @ 1024x256)
// =====================================================================
__global__ __launch_bounds__(256) void hproj_kernel(const float* __restrict__ scene,
                                                    const float* __restrict__ W,
                                                    const float* __restrict__ b,
                                                    float* __restrict__ h) {
    __shared__ float srow[1024];
    int rowi = blockIdx.x, tid = threadIdx.x;
#pragma unroll
    for (int q = 0; q < 4; ++q) srow[q * 256 + tid] = scene[(size_t)rowi * 1024 + q * 256 + tid];
    __syncthreads();
    float acc = b[tid];
    for (int k = 0; k < 1024; ++k) acc += srow[k] * W[k * 256 + tid];
    h[rowi * 256 + tid] = acc;
}

// =====================================================================
// Mamba: fused rmsnorm + in_proj -> hg (row x 1024; [0:512]=h, [512:1024]=gate)
// =====================================================================
__global__ __launch_bounds__(256) void mamba_in_kernel(const float* __restrict__ h,
                                                       const float* __restrict__ nw,
                                                       const float* __restrict__ W,
                                                       float* __restrict__ hg) {
    __shared__ float xr[256];
    __shared__ float red[256];
    int rowi = blockIdx.x, tid = threadIdx.x;
    float hv = h[rowi * 256 + tid];
    red[tid] = hv * hv;
    __syncthreads();
    for (int o = 128; o > 0; o >>= 1) { if (tid < o) red[tid] += red[tid + o]; __syncthreads(); }
    float rms = rsqrtf(red[0] * (1.0f / 256.0f) + 1e-5f);
    xr[tid] = hv * rms * nw[tid];
    __syncthreads();
#pragma unroll
    for (int q = 0; q < 4; ++q) {
        int c = q * 256 + tid;
        float acc = 0.0f;
        for (int k = 0; k < 256; ++k) acc += xr[k] * W[k * 1024 + c];
        hg[(size_t)rowi * 1024 + c] = acc;
    }
}

// =====================================================================
// Mamba: causal depthwise conv (K=4) + bias + SiLU -> ht
// =====================================================================
__global__ __launch_bounds__(256) void conv_kernel(const float* __restrict__ hg,
                                                   const float* __restrict__ cw,
                                                   const float* __restrict__ cb,
                                                   float* __restrict__ ht) {
    int rowi = blockIdx.x;
    int b = rowi >> 9, t = rowi & 511;
    int tid = threadIdx.x;
#pragma unroll
    for (int q = 0; q < 2; ++q) {
        int d = q * 256 + tid;
        float acc = cb[d];
#pragma unroll
        for (int k = 0; k < 4; ++k) {
            int tt = t - 3 + k;
            if (tt >= 0) acc += cw[d * 4 + k] * hg[(size_t)(b * 512 + tt) * 1024 + d];
        }
        ht[(size_t)rowi * 512 + d] = acc * sigmoidf_(acc);
    }
}

// =====================================================================
// Mamba: ssm = ht @ x_proj ; B,C split ; dt = softplus(ssm[:16]@dt_w + dt_b)
// =====================================================================
__global__ __launch_bounds__(256) void xproj_kernel(const float* __restrict__ ht,
                                                    const float* __restrict__ xp,
                                                    const float* __restrict__ dtw,
                                                    const float* __restrict__ dtb,
                                                    float* __restrict__ Bb,
                                                    float* __restrict__ Cb,
                                                    float* __restrict__ dtv) {
    __shared__ float hr[512];
    __shared__ float ssm[48];
    int rowi = blockIdx.x, tid = threadIdx.x;
    hr[tid]       = ht[(size_t)rowi * 512 + tid];
    hr[256 + tid] = ht[(size_t)rowi * 512 + 256 + tid];
    __syncthreads();
    if (tid < 48) {
        float acc = 0.0f;
        for (int k = 0; k < 512; ++k) acc += hr[k] * xp[k * 48 + tid];
        ssm[tid] = acc;
    }
    __syncthreads();
    if (tid < 16) Bb[rowi * 16 + tid] = ssm[16 + tid];
    else if (tid < 32) Cb[rowi * 16 + (tid - 16)] = ssm[32 + (tid - 16)];
#pragma unroll
    for (int q = 0; q < 2; ++q) {
        int c = q * 256 + tid;
        float acc = dtb[c];
#pragma unroll
        for (int r2 = 0; r2 < 16; ++r2) acc += ssm[r2] * dtw[r2 * 512 + c];
        dtv[(size_t)rowi * 512 + c] = softplusf_(acc);
    }
}

// =====================================================================
// Mamba: selective scan over L=512 (4 blocks, 512 channels each), + skip D,
// + gate SiLU -> ym
// =====================================================================
__global__ __launch_bounds__(512) void scan_kernel(const float* __restrict__ dtv,
                                                   const float* __restrict__ htb,
                                                   const float* __restrict__ Bb,
                                                   const float* __restrict__ Cb,
                                                   const float* __restrict__ A_log,
                                                   const float* __restrict__ Dp,
                                                   const float* __restrict__ hg,
                                                   float* __restrict__ ym) {
    __shared__ float lB[16], lC[16];
    int b = blockIdx.x, d = threadIdx.x;
    float a[16], st[16];
#pragma unroll
    for (int n = 0; n < 16; ++n) { a[n] = -__expf(A_log[d * 16 + n]); st[n] = 0.0f; }
    float Dd = Dp[d];
    for (int t = 0; t < 512; ++t) {
        int rowi = b * 512 + t;
        __syncthreads();
        if (d < 16) lB[d] = Bb[rowi * 16 + d];
        else if (d < 32) lC[d - 16] = Cb[rowi * 16 + (d - 16)];
        __syncthreads();
        float dv = dtv[(size_t)rowi * 512 + d];
        float hv = htb[(size_t)rowi * 512 + d];
        float sc = dv * hv;
        float y = 0.0f;
#pragma unroll
        for (int n = 0; n < 16; ++n) {
            float dA = __expf(dv * a[n]);
            st[n] = dA * st[n] + sc * lB[n];
            y += st[n] * lC[n];
        }
        y += hv * Dd;
        float g = hg[(size_t)rowi * 1024 + 512 + d];
        y *= g * sigmoidf_(g);
        ym[(size_t)rowi * 512 + d] = y;
    }
}

// =====================================================================
// Mamba: h += ym @ out_proj  (residual)
// =====================================================================
__global__ __launch_bounds__(256) void outproj_kernel(const float* __restrict__ ym,
                                                      const float* __restrict__ W,
                                                      float* __restrict__ h) {
    __shared__ float yr[512];
    int rowi = blockIdx.x, tid = threadIdx.x;
    yr[tid]       = ym[(size_t)rowi * 512 + tid];
    yr[256 + tid] = ym[(size_t)rowi * 512 + 256 + tid];
    __syncthreads();
    float acc = 0.0f;
    for (int k = 0; k < 512; ++k) acc += yr[k] * W[k * 256 + tid];
    h[rowi * 256 + tid] += acc;
}

// =====================================================================
// final rmsnorm
// =====================================================================
__global__ __launch_bounds__(256) void rms_kernel(const float* __restrict__ h,
                                                  const float* __restrict__ nw,
                                                  float* __restrict__ hn) {
    __shared__ float red[256];
    int rowi = blockIdx.x, tid = threadIdx.x;
    float hv = h[rowi * 256 + tid];
    red[tid] = hv * hv;
    __syncthreads();
    for (int o = 128; o > 0; o >>= 1) { if (tid < o) red[tid] += red[tid + o]; __syncthreads(); }
    hn[rowi * 256 + tid] = hv * rsqrtf(red[0] * (1.0f / 256.0f) + 1e-5f) * nw[tid];
}

// =====================================================================
// episodic memory: precompute Kn (normalized keys) and V from memory slots
// =====================================================================
__global__ __launch_bounds__(256) void mem_kv_kernel(const float* __restrict__ mem,
                                                     const float* __restrict__ mkw,
                                                     const float* __restrict__ mkb,
                                                     const float* __restrict__ mvw,
                                                     const float* __restrict__ mvb,
                                                     float* __restrict__ Kn,
                                                     float* __restrict__ Vv) {
    __shared__ float mr[256];
    __shared__ float red[256];
    int m = blockIdx.x, tid = threadIdx.x;
    mr[tid] = mem[m * 256 + tid];
    __syncthreads();
    float kacc = mkb[tid], vacc = mvb[tid];
    for (int k = 0; k < 256; ++k) {
        float mv = mr[k];
        kacc += mv * mkw[k * 256 + tid];
        vacc += mv * mvw[k * 256 + tid];
    }
    Vv[m * 256 + tid] = vacc;
    red[tid] = kacc * kacc;
    __syncthreads();
    for (int o = 128; o > 0; o >>= 1) { if (tid < o) red[tid] += red[tid + o]; __syncthreads(); }
    float nrm = fmaxf(sqrtf(red[0]), 1e-12f);
    Kn[m * 256 + tid] = kacc / nrm;
}

// =====================================================================
// episodic memory: Q, cosine attention over 50 slots, retrieval, forget gate
// =====================================================================
__global__ __launch_bounds__(256) void epmem_kernel(const float* __restrict__ hn,
                                                    const float* __restrict__ mqw,
                                                    const float* __restrict__ mqb,
                                                    const float* __restrict__ Kn,
                                                    const float* __restrict__ Vv,
                                                    const float* __restrict__ fgw,
                                                    const float* __restrict__ fgb,
                                                    float* __restrict__ out) {
    __shared__ float hr[256];
    __shared__ float Q[256];
    __shared__ float att[50];
    __shared__ float R[256];
    __shared__ float red[256];
    __shared__ float stat[1];
    int rowi = blockIdx.x, tid = threadIdx.x;
    float hv = hn[rowi * 256 + tid];
    hr[tid] = hv;
    __syncthreads();
    float q = mqb[tid];
    for (int k = 0; k < 256; ++k) q += hr[k] * mqw[k * 256 + tid];
    Q[tid] = q;
    red[tid] = q * q;
    __syncthreads();
    for (int o = 128; o > 0; o >>= 1) { if (tid < o) red[tid] += red[tid + o]; __syncthreads(); }
    float qs = 1.0f / fmaxf(sqrtf(red[0]), 1e-12f);
    if (tid < 50) {
        float acc = 0.0f;
        for (int k = 0; k < 256; ++k) acc += Q[k] * Kn[tid * 256 + k];
        att[tid] = acc * qs * (1.0f / 16.0f);
    }
    __syncthreads();
    if (tid == 0) {
        float mx = att[0];
        for (int m2 = 1; m2 < 50; ++m2) mx = fmaxf(mx, att[m2]);
        float se = 0.0f;
        for (int m2 = 0; m2 < 50; ++m2) { float e = __expf(att[m2] - mx); att[m2] = e; se += e; }
        stat[0] = 1.0f / se;
    }
    __syncthreads();
    float inv = stat[0];
    float racc = 0.0f;
    for (int m2 = 0; m2 < 50; ++m2) racc += att[m2] * Vv[m2 * 256 + tid];
    racc *= inv;
    R[tid] = racc;
    __syncthreads();
    float f = fgb[tid];
    for (int k = 0; k < 256; ++k) f += hr[k] * fgw[k * 256 + tid];
    for (int k = 0; k < 256; ++k) f += R[k] * fgw[(256 + k) * 256 + tid];
    f = sigmoidf_(f);
    out[(size_t)rowi * 256 + tid] = f * hv + (1.0f - f) * racc;
}

// =====================================================================
// host launcher
// =====================================================================
extern "C" void kernel_launch(void* const* d_in, const int* in_sizes, int n_in,
                              void* d_out, int out_size, void* d_ws, size_t ws_size,
                              hipStream_t stream) {
    (void)in_sizes; (void)n_in; (void)out_size; (void)ws_size;

    const float* X            = (const float*)d_in[0];
    const unsigned char* mask = (const unsigned char*)d_in[1];   // jax bool -> 1 byte/elem
    const float* sg_w1  = (const float*)d_in[2];
    const float* sg_b1  = (const float*)d_in[3];
    const float* sg_w2  = (const float*)d_in[4];
    const float* sg_b2  = (const float*)d_in[5];
    const float* declam = (const float*)d_in[6];
    const float* query  = (const float*)d_in[7];
    const float* key_w  = (const float*)d_in[8];
    const float* key_b  = (const float*)d_in[9];
    const float* proj_w = (const float*)d_in[10];
    const float* proj_b = (const float*)d_in[11];
    const float* norm_f = (const float*)d_in[12];
    const float* memory = (const float*)d_in[13];
    const float* mq_w   = (const float*)d_in[14];
    const float* mq_b   = (const float*)d_in[15];
    const float* mk_w   = (const float*)d_in[16];
    const float* mk_b   = (const float*)d_in[17];
    const float* mv_w   = (const float*)d_in[18];
    const float* mv_b   = (const float*)d_in[19];
    const float* fg_w   = (const float*)d_in[20];
    const float* fg_b   = (const float*)d_in[21];

    // workspace layout (floats)
    float* fp = (float*)d_ws;
    float* wkq    = fp;                 fp += 1280;       // kq[1024] + kb.q at [1024]
    float* wS     = fp;                 fp += NTOK;       // gate s
    float* wDot   = fp;                 fp += NTOK;       // x . kq
    float* wW     = fp;                 fp += NTOK;       // alpha * s
    float* wScene = fp;                 fp += (size_t)NROW * DM1;
    float* wH     = fp;                 fp += (size_t)NROW * DD;
    float* wHN    = fp;                 fp += (size_t)NROW * DD;
    float* wHG    = fp;                 fp += (size_t)NROW * 2 * DINNER;
    float* wHT    = fp;                 fp += (size_t)NROW * DINNER;
    float* wDT    = fp;                 fp += (size_t)NROW * DINNER;
    float* wB     = fp;                 fp += (size_t)NROW * DSTATE;
    float* wC     = fp;                 fp += (size_t)NROW * DSTATE;
    float* wY     = fp;                 fp += (size_t)NROW * DINNER;
    float* wKn    = fp;                 fp += NSLOTS * DD;
    float* wV     = fp;                 fp += NSLOTS * DD;
    unsigned short* wBswz = (unsigned short*)fp;          // 524288 bf16

    // --- scene pool ---
    swz_w1_kernel<<<2048, 256, 0, stream>>>(sg_w1, wBswz);
    kq_kernel<<<5, 256, 0, stream>>>(key_w, key_b, query, wkq);
    gemm_gate_kernel<<<NTOK / 32, 256, 0, stream>>>(X, wBswz, sg_b1, sg_w2, sg_b2, wkq, wS, wDot);
    pool_softmax_kernel<<<NROW, 32, 0, stream>>>(wS, wDot, mask, wkq, declam, wW);
    scene_sum_kernel<<<NROW, 256, 0, stream>>>(X, wW, wScene);
    hproj_kernel<<<NROW, 256, 0, stream>>>(wScene, proj_w, proj_b, wH);

    // --- mamba layers ---
    for (int l = 0; l < 4; ++l) {
        const int base = 22 + l * 10;
        const float* lnorm    = (const float*)d_in[base + 0];
        const float* in_proj  = (const float*)d_in[base + 1];
        const float* conv_w   = (const float*)d_in[base + 2];
        const float* conv_b   = (const float*)d_in[base + 3];
        const float* x_proj   = (const float*)d_in[base + 4];
        const float* dt_w     = (const float*)d_in[base + 5];
        const float* dt_b     = (const float*)d_in[base + 6];
        const float* A_log    = (const float*)d_in[base + 7];
        const float* Dp       = (const float*)d_in[base + 8];
        const float* out_proj = (const float*)d_in[base + 9];

        mamba_in_kernel<<<NROW, 256, 0, stream>>>(wH, lnorm, in_proj, wHG);
        conv_kernel<<<NROW, 256, 0, stream>>>(wHG, conv_w, conv_b, wHT);
        xproj_kernel<<<NROW, 256, 0, stream>>>(wHT, x_proj, dt_w, dt_b, wB, wC, wDT);
        scan_kernel<<<NBS, 512, 0, stream>>>(wDT, wHT, wB, wC, A_log, Dp, wHG, wY);
        outproj_kernel<<<NROW, 256, 0, stream>>>(wY, out_proj, wH);
    }

    // --- final norm + episodic memory ---
    rms_kernel<<<NROW, 256, 0, stream>>>(wH, norm_f, wHN);
    mem_kv_kernel<<<NSLOTS, 256, 0, stream>>>(memory, mk_w, mk_b, mv_w, mv_b, wKn, wV);
    epmem_kernel<<<NROW, 256, 0, stream>>>(wHN, mq_w, mq_b, wKn, wV, fg_w, fg_b, (float*)d_out);
}
